// LatentPositionModel_48017734369849
// MI455X (gfx1250) — matrix operand order repaired and checked by reading, not compiled, not run on hardware
//
#include <hip/hip_runtime.h>

#define D_DIM 128

typedef __attribute__((ext_vector_type(16))) _Float16 v16h;
typedef __attribute__((ext_vector_type(8)))  _Float16 v8h;
typedef __attribute__((ext_vector_type(4)))  _Float16 v4h;
typedef __attribute__((ext_vector_type(8)))  float    v8f;
typedef __attribute__((ext_vector_type(4)))  float    v4f;

__device__ __forceinline__ float wave_sum32(float v) {
#pragma unroll
  for (int off = 16; off > 0; off >>= 1)
    v += __shfl_xor(v, off, 32);
  return v;
}

// Load 8 consecutive f16 elements starting at column byte-offset c (elements).
// USE_H: one b128 load from the L2-resident f16 table.
// !USE_H: two b128 f32 loads + pack-convert.
template <bool USE_H>
__device__ __forceinline__ v8h load8(const float* __restrict__ pf,
                                     const _Float16* __restrict__ ph, int c) {
  if (USE_H) {
    return *(const v8h*)(ph + c);
  } else {
    v4f x = *(const v4f*)(pf + c);
    v4f y = *(const v4f*)(pf + c + 4);
    v8h h;
    h[0] = (_Float16)x[0]; h[1] = (_Float16)x[1];
    h[2] = (_Float16)x[2]; h[3] = (_Float16)x[3];
    h[4] = (_Float16)y[0]; h[5] = (_Float16)y[1];
    h[6] = (_Float16)y[2]; h[7] = (_Float16)y[3];
    return h;
  }
}

// One wave per row: f32->f16 cast of mu, row squared-norms, and the KL term.
template <bool USE_H>
__global__ void lpm_prep_kernel(const float* __restrict__ mu,
                                const float* __restrict__ log_sigma,
                                _Float16* __restrict__ muh,
                                float* __restrict__ norm2,
                                float* __restrict__ partials,
                                int N) {
  const int lane = threadIdx.x & 31;
  const int wib  = threadIdx.x >> 5;
  const int wpb  = blockDim.x >> 5;
  long long wave  = (long long)blockIdx.x * wpb + wib;
  long long nwave = (long long)gridDim.x * wpb;

  float kl = 0.f;
  for (long long r = wave; r < N; r += nwave) {
    // streamed once -> non-temporal
    v4f m  = __builtin_nontemporal_load((const v4f*)(mu        + (size_t)r * D_DIM + lane * 4));
    v4f ls = __builtin_nontemporal_load((const v4f*)(log_sigma + (size_t)r * D_DIM + lane * 4));
    float np = 0.f;
#pragma unroll
    for (int i = 0; i < 4; ++i) {
      float mi = m[i], li = ls[i];
      np += mi * mi;
      // 0.5 * (sigma^2 + mu^2 - log_sigma - 1), sigma^2 = exp(2*log_sigma)
      kl += 0.5f * (__expf(2.f * li) + mi * mi - li - 1.f);
    }
    if (USE_H) {
      v4h h;
      h[0] = (_Float16)m[0]; h[1] = (_Float16)m[1];
      h[2] = (_Float16)m[2]; h[3] = (_Float16)m[3];
      // regular store: table must stay L2-resident for the gather phase
      *(v4h*)(muh + (size_t)r * D_DIM + lane * 4) = h;
    }
    np = wave_sum32(np);
    if (lane == 0) norm2[r] = np;
  }

  __shared__ float smem[32];
  kl = wave_sum32(kl);
  if (lane == 0) smem[wib] = kl;
  __syncthreads();
  if (threadIdx.x == 0) {
    float s = 0.f;
    for (int i = 0; i < wpb; ++i) s += smem[i];
    partials[blockIdx.x] = s;
  }
}

// One wave per tile of 16 edges. WMMA computes the 16x16 cross-dot matrix of
// src rows (A) x dst rows (B); the diagonal holds the 16 edge dot products.
// d2_e = |src|^2 + |dst|^2 - 2*dot_e ; contribution = -log_sigmoid(-d2)
//       = d2 + log1p(exp(-d2)).
template <bool USE_H>
__global__ void lpm_edge_kernel(const float* __restrict__ mu,
                                const _Float16* __restrict__ muh,
                                const long long* __restrict__ ei,
                                const float* __restrict__ norm2,
                                float* __restrict__ partials,
                                long long E) {
  const int lane = threadIdx.x & 31;
  const int wib  = threadIdx.x >> 5;
  const int wpb  = blockDim.x >> 5;
  const int hv   = lane >> 4;   // lane half: 0 = lanes 0-15, 1 = lanes 16-31
  const int col  = lane & 15;   // A row index M / B column index N / edge slot
  long long wave  = (long long)blockIdx.x * wpb + wib;
  long long nwave = (long long)gridDim.x * wpb;
  long long ntile = (E + 15) >> 4;

  float acc = 0.f;

  long long t = wave;
  long long s = 0, d = 0, e = 0;
  if (t < ntile) {  // prologue: load first tile's indices (NT: streamed once)
    e = t * 16 + col;
    long long ec = (e < E) ? e : (E - 1);
    s = __builtin_nontemporal_load(ei + ec);
    d = __builtin_nontemporal_load(ei + E + ec);
  }

  while (t < ntile) {
    const long long s_cur = s, d_cur = d, e_cur = e;
    const long long tn = t + nwave;
    if (tn < ntile) {  // issue next tile's index loads before the gather work
      e = tn * 16 + col;
      long long ec = (e < E) ? e : (E - 1);
      s = __builtin_nontemporal_load(ei + ec);
      d = __builtin_nontemporal_load(ei + E + ec);
    }

    // hoisted row base pointers -> gathers become base + immediate offsets
    const _Float16* __restrict__ phs = muh + (size_t)s_cur * D_DIM;
    const _Float16* __restrict__ phd = muh + (size_t)d_cur * D_DIM;
    const float*    __restrict__ pfs = mu  + (size_t)s_cur * D_DIM;
    const float*    __restrict__ pfd = mu  + (size_t)d_cur * D_DIM;

    v8f c = {};
#pragma unroll
    for (int k = 0; k < D_DIM; k += 32) {
      // A fragment (16x32 f16): lane {hv,M=col}: K {hv*8..+7}, {16+hv*8..+7}
      v8h a0 = load8<USE_H>(pfs, phs, k + hv * 8);
      v8h a1 = load8<USE_H>(pfs, phs, k + 16 + hv * 8);
      // B fragment (32x16 f16): lane {hv,N=col}: K {hv*8..+7}, {16+hv*8..+7}
      v8h b0 = load8<USE_H>(pfd, phd, k + hv * 8);
      v8h b1 = load8<USE_H>(pfd, phd, k + 16 + hv * 8);
      v16h a = __builtin_shufflevector(a0, a1, 0,1,2,3,4,5,6,7,8,9,10,11,12,13,14,15);
      v16h b = __builtin_shufflevector(b0, b1, 0,1,2,3,4,5,6,7,8,9,10,11,12,13,14,15);
      c = __builtin_amdgcn_wmma_f32_16x16x32_f16(
          /*neg_a=*/false, a, /*neg_b=*/false, b,
          /*c_mod=*/(short)0, c, /*reuse_a=*/false, /*reuse_b=*/false);
    }

    // Diagonal extraction: element (m,n) lives in vgpr m%8, lane (m/8)*16 + n.
    // This lane holds diagonal element `col` iff j + hv*8 == col.
    float dot = 0.f; bool have = false;
#pragma unroll
    for (int j = 0; j < 8; ++j) {
      if (j + hv * 8 == col) { dot = c[j]; have = true; }
    }
    if (have && e_cur < E) {
      float d2 = norm2[s_cur] + norm2[d_cur] - 2.f * dot;
      float z  = __expf(-d2);
      acc += d2 + __logf(1.f + z);
    }
    t = tn;
  }

  __shared__ float smem[32];
  acc = wave_sum32(acc);
  if (lane == 0) smem[wib] = acc;
  __syncthreads();
  if (threadIdx.x == 0) {
    float sblk = 0.f;
    for (int i = 0; i < wpb; ++i) sblk += smem[i];
    partials[blockIdx.x] = sblk;
  }
}

// Deterministic final reduction of all block partials -> scalar output.
__global__ void lpm_finalize_kernel(const float* __restrict__ partials, int n,
                                    float* __restrict__ out) {
  __shared__ float smem[256];
  float s = 0.f;
  for (int i = threadIdx.x; i < n; i += blockDim.x) s += partials[i];
  smem[threadIdx.x] = s;
  __syncthreads();
  for (int off = blockDim.x >> 1; off > 0; off >>= 1) {
    if ((int)threadIdx.x < off) smem[threadIdx.x] += smem[threadIdx.x + off];
    __syncthreads();
  }
  if (threadIdx.x == 0) out[0] = smem[0];
}

extern "C" void kernel_launch(void* const* d_in, const int* in_sizes, int n_in,
                              void* d_out, int out_size, void* d_ws, size_t ws_size,
                              hipStream_t stream) {
  const float*     mu        = (const float*)d_in[0];
  const float*     log_sigma = (const float*)d_in[1];
  const long long* ei        = (const long long*)d_in[2];  // int64 [2, E]

  const int       N = in_sizes[0] / D_DIM;
  const long long E = (long long)in_sizes[2] / 2;

  const int EDGE_BLOCKS = 4096;
  const int PREP_BLOCKS = 2048;
  const int TPB         = 256;

  // Workspace layout: [edge partials][prep partials][norm2][f16 mu table]
  char*  ws           = (char*)d_ws;
  float* partialsEdge = (float*)ws;                                   // contiguous with
  float* partialsPrep = (float*)(ws + (size_t)EDGE_BLOCKS * 4);       // edge partials
  size_t off = (size_t)(EDGE_BLOCKS + PREP_BLOCKS) * 4;
  off = (off + 255) & ~(size_t)255;
  float* norm2 = (float*)(ws + off);
  off += (size_t)N * 4;
  off = (off + 255) & ~(size_t)255;
  _Float16* muh = (_Float16*)(ws + off);
  const size_t need_f16 = off + (size_t)N * D_DIM * 2;
  const bool use_h = (ws_size >= need_f16);   // fallback: gather f32 directly

  if (use_h) {
    lpm_prep_kernel<true><<<PREP_BLOCKS, TPB, 0, stream>>>(mu, log_sigma, muh,
                                                           norm2, partialsPrep, N);
    lpm_edge_kernel<true><<<EDGE_BLOCKS, TPB, 0, stream>>>(mu, muh, ei, norm2,
                                                           partialsEdge, E);
  } else {
    lpm_prep_kernel<false><<<PREP_BLOCKS, TPB, 0, stream>>>(mu, log_sigma, muh,
                                                            norm2, partialsPrep, N);
    lpm_edge_kernel<false><<<EDGE_BLOCKS, TPB, 0, stream>>>(mu, muh, ei, norm2,
                                                            partialsEdge, E);
  }
  lpm_finalize_kernel<<<1, TPB, 0, stream>>>(partialsEdge,
                                             EDGE_BLOCKS + PREP_BLOCKS,
                                             (float*)d_out);
}
